// Model_84172769067909
// MI455X (gfx1250) — compile-verified, matrix operand order checked by reading, not run on hardware
//
#include <hip/hip_runtime.h>
#include <hip/hip_bf16.h>
#include <math.h>

// ---------------------------------------------------------------------------
// LAS encoder-decoder on MI455X (gfx1250), wave32 + v_wmma_f32_16x16x32_bf16.
// GEMM operands pre-packed to zero-padded bf16: each WMMA fragment is two
// unguarded 16B loads.  Batched GEMM uses 2x2 register tiles per wave
// (4 WMMA per 8 b128 loads).  f32 kept for gate accumulators.
// Workspace requirement: ~460 MB.
// ---------------------------------------------------------------------------

typedef __attribute__((ext_vector_type(16))) __bf16 v16bf;
typedef __attribute__((ext_vector_type(8)))  __bf16 v8bf;
typedef __attribute__((ext_vector_type(8)))  float  v8f;

// Fragment (A or B operand) of v_wmma_f32_16x16x32_bf16 from a padded bf16
// matrix (row-major, ld multiple of 8, K region zero-padded, rows valid).
// Per ISA 7.12.2: lane holds row = lane%16; lanes<16 carry K octets {0..7,
// 16..23}, lanes>=16 carry {8..15,24..31}  ->  two contiguous 8-elem chunks.
__device__ __forceinline__ v16bf frag_ld(const __bf16* base, int row0, int ld,
                                         int k0, int lane) {
    const __bf16* p = base + (size_t)(row0 + (lane & 15)) * ld + k0 +
                      ((lane & 16) ? 8 : 0);
    v8bf lo = *(const v8bf*)(p);
    v8bf hi = *(const v8bf*)(p + 16);
    return __builtin_shufflevector(lo, hi, 0, 1, 2, 3, 4, 5, 6, 7,
                                   8, 9, 10, 11, 12, 13, 14, 15);
}

__device__ __forceinline__ float sigm(float x) { return 1.0f / (1.0f + __expf(-x)); }

// ---------------------------------------------------------------------------
// Pack f32 sub-matrix -> padded bf16: dst[r][dstOff+k] (r<Rpad, k<Kc) =
// (r<R && k<K) ? src[r*ldSrc + colOff + k] : 0
// ---------------------------------------------------------------------------
__global__ void pack_bf16_kernel(const float* __restrict__ src, int ldSrc,
                                 int colOff, int R, int K,
                                 __bf16* __restrict__ dst, int ldDst,
                                 int dstOff, int Rpad, int Kc) {
    int total = Rpad * Kc;
    for (int idx = blockIdx.x * blockDim.x + threadIdx.x; idx < total;
         idx += gridDim.x * blockDim.x) {
        int r = idx / Kc, k = idx - r * Kc;
        float v = (r < R && k < K) ? src[(size_t)r * ldSrc + colOff + k] : 0.0f;
        dst[(size_t)r * ldDst + dstOff + k] = (__bf16)v;
    }
}

__global__ void fill_zero_bf16_kernel(__bf16* __restrict__ p, size_t n) {
    for (size_t i = blockIdx.x * (size_t)blockDim.x + threadIdx.x; i < n;
         i += gridDim.x * (size_t)blockDim.x)
        p[i] = (__bf16)0.0f;
}

// ---------------------------------------------------------------------------
// GEMM: C[M,N](f32) = A[M,Kpad](bf16) @ W[Npad,Kpad](bf16)^T + b0 + b1.
// Block = 256 thr = 8 waves arranged 2(M) x 4(N); each wave owns a 32x32
// output tile = 2x2 WMMA accumulators (4 WMMA per 8 b128 loads).
// Block tile 64(M) x 128(N).  M multiple of 64; W rows padded to
// gridDim.x*128; K32 = Kpad/32.
// ---------------------------------------------------------------------------
__global__ __launch_bounds__(256) void gemm_bias_kernel(
    const __bf16* __restrict__ A, int lda,
    const __bf16* __restrict__ W, int ldw,
    const float* __restrict__ b0, const float* __restrict__ b1,
    float* __restrict__ C, int ldc, int M, int N, int K32) {
    int tid = threadIdx.x, lane = tid & 31, wave = tid >> 5;
    int m0 = blockIdx.y * 64 + (wave & 1) * 32;
    int n0 = blockIdx.x * 128 + (wave >> 1) * 32;
    v8f acc00 = {}, acc01 = {}, acc10 = {}, acc11 = {};
    for (int kt = 0; kt < K32; ++kt) {
        v16bf a0 = frag_ld(A, m0, lda, kt * 32, lane);
        v16bf a1 = frag_ld(A, m0 + 16, lda, kt * 32, lane);
        v16bf w0 = frag_ld(W, n0, ldw, kt * 32, lane);
        v16bf w1 = frag_ld(W, n0 + 16, ldw, kt * 32, lane);
        acc00 = __builtin_amdgcn_wmma_f32_16x16x32_bf16(false, a0, false, w0,
                                                        (short)0, acc00, false, false);
        acc01 = __builtin_amdgcn_wmma_f32_16x16x32_bf16(false, a0, false, w1,
                                                        (short)0, acc01, false, false);
        acc10 = __builtin_amdgcn_wmma_f32_16x16x32_bf16(false, a1, false, w0,
                                                        (short)0, acc10, false, false);
        acc11 = __builtin_amdgcn_wmma_f32_16x16x32_bf16(false, a1, false, w1,
                                                        (short)0, acc11, false, false);
    }
#pragma unroll
    for (int mi = 0; mi < 2; ++mi) {
#pragma unroll
        for (int ni = 0; ni < 2; ++ni) {
            v8f acc = (mi == 0) ? (ni == 0 ? acc00 : acc01)
                                : (ni == 0 ? acc10 : acc11);
            int n = n0 + ni * 16 + (lane & 15);
            if (n < N) {
                float bias = (b0 ? b0[n] : 0.0f) + (b1 ? b1[n] : 0.0f);
                int mbase = m0 + mi * 16 + ((lane & 16) ? 8 : 0);
#pragma unroll
                for (int r = 0; r < 8; ++r) {
                    int m = mbase + r;
                    if (m < M) C[(size_t)m * ldc + n] = acc[r] + bias;
                }
            }
        }
    }
}

// ---------------------------------------------------------------------------
// Persistent BiLSTM recurrence.  Grid = 2 (fwd/bwd), block = 1024 = 32 waves.
// h in LDS bf16 [32 x 320] (zero-padded K).  Per step: 150 WMMA tiles of
// gates = h @ Whh^T (+ xproj[t] f32), staged via L2 scratch, fused pointwise,
// c in registers, h written bf16 to out [T,32,608] (pad cols pre-zeroed).
// ---------------------------------------------------------------------------
__global__ __launch_bounds__(1024) void lstm_seq_kernel(
    const float* __restrict__ xproj, size_t xproj_dir_stride,
    const __bf16* __restrict__ Whh_f, const __bf16* __restrict__ Whh_b,
    __bf16* __restrict__ out,         // [T,32,608]; dir d -> cols d*300..
    float* __restrict__ gates_ws,     // [2][32*1200] f32 scratch (L2)
    int T) {
    int dir = blockIdx.x;
    const float*  xp  = xproj + (size_t)dir * xproj_dir_stride;
    const __bf16* Whh = dir ? Whh_b : Whh_f;
    float* gates = gates_ws + (size_t)dir * (32 * 1200);

    __shared__ __bf16 h_lds[32 * 320];
    int tid = threadIdx.x, lane = tid & 31, wave = tid >> 5;
    for (int i = tid; i < 32 * 320; i += 1024) h_lds[i] = (__bf16)0.0f;
    float c[10];
#pragma unroll
    for (int j = 0; j < 10; ++j) c[j] = 0.0f;
    __syncthreads();

    for (int s = 0; s < T; ++s) {
        int t = dir ? (T - 1 - s) : s;
        const float* xpt = xp + (size_t)t * 32 * 1200;
        for (int tt = wave; tt < 150; tt += 32) {
            int m0 = (tt & 1) * 16, n0 = (tt >> 1) * 16;
            v8f acc = {};
#pragma unroll
            for (int kt = 0; kt < 10; ++kt) {
                v16bf a = frag_ld(h_lds, m0, 320, kt * 32, lane);
                v16bf b = frag_ld(Whh, n0, 320, kt * 32, lane);
                acc = __builtin_amdgcn_wmma_f32_16x16x32_bf16(
                    false, a, false, b, (short)0, acc, false, false);
            }
            int n = n0 + (lane & 15);
            int mbase = m0 + ((lane & 16) ? 8 : 0);
#pragma unroll
            for (int r = 0; r < 8; ++r) {
                int m = mbase + r;
                gates[m * 1200 + n] = acc[r] + xpt[m * 1200 + n];
            }
        }
        __threadfence();
        __syncthreads();
        __bf16* outt = out + (size_t)t * 32 * 608 + dir * 300;
#pragma unroll
        for (int j = 0; j < 10; ++j) {
            int i = tid + j * 1024;
            if (i < 9600) {
                int b = i / 300, e = i - b * 300;
                const float* g = gates + b * 1200;
                float ig = sigm(g[e]);
                float fg = sigm(g[300 + e]);
                float gg = tanhf(g[600 + e]);
                float og = sigm(g[900 + e]);
                float cn = fg * c[j] + ig * gg;
                c[j] = cn;
                float hn = og * tanhf(cn);
                h_lds[b * 320 + e] = (__bf16)hn;
                outt[b * 608 + e] = (__bf16)hn;
            }
        }
        __syncthreads();
    }
}

// Pyramid repack: bf16 [T,32,608] -> bf16 [T/2,32,1216] (cols>=1200 zeroed)
__global__ void pyramid_kernel(const __bf16* __restrict__ in,
                               __bf16* __restrict__ out, int Thalf) {
    int total = Thalf * 32 * 1216;
    for (int idx = blockIdx.x * blockDim.x + threadIdx.x; idx < total;
         idx += gridDim.x * blockDim.x) {
        int s = idx / (32 * 1216);
        int r = idx - s * (32 * 1216);
        int b = r / 1216, e = r - b * 1216;
        __bf16 v = (__bf16)0.0f;
        if (e < 1200) {
            int t = 2 * s + e / 600, d = e % 600;
            v = in[((size_t)t * 32 + b) * 608 + d];
        }
        out[idx] = v;
    }
}

// Embedding gather -> padded bf16 [6400 x 416]
__global__ void embed_kernel(const int* __restrict__ Y,
                             const float* __restrict__ emb,
                             __bf16* __restrict__ embX) {
    int total = 6400 * 416;
    for (int idx = blockIdx.x * blockDim.x + threadIdx.x; idx < total;
         idx += gridDim.x * blockDim.x) {
        int row = idx / 416, k = idx - row * 416;
        float v = (k < 400) ? emb[(size_t)Y[row] * 400 + k] : 0.0f;
        embX[idx] = (__bf16)v;
    }
}

// ---------------------------------------------------------------------------
// Fused persistent decoder: one block of 32 waves loops u = 0..199.
// hc (bf16 LDS, ld 448) = [h(300) | pad(20) | ctx(128)]; c in registers.
// decW (bf16, [1200 x 448]) = [Whh | pad | Wih_ctx] matches hc's K layout.
// ---------------------------------------------------------------------------
__global__ __launch_bounds__(1024) void decoder_kernel(
    const float* __restrict__ xprojdec,   // [200*32,1200] f32
    const float* __restrict__ keys,       // [128,32,128] f32
    const float* __restrict__ values,     // [128,32,128] f32
    const __bf16* __restrict__ phiW,      // [128 x 320] bf16 padded
    const float* __restrict__ phib,
    const __bf16* __restrict__ decW,      // [1200 x 448] bf16
    __bf16* __restrict__ lc,              // [6400 x 448] bf16 (pad pre-zeroed)
    float* __restrict__ gates) {          // [32*1200] f32 scratch
    __shared__ __bf16 hc[32 * 448];
    __shared__ float qs[32 * 128];
    __shared__ float sc[32 * 128];
    __shared__ float ctx[32 * 128];
    int tid = threadIdx.x, lane = tid & 31, wave = tid >> 5;
    for (int i = tid; i < 32 * 448; i += 1024) hc[i] = (__bf16)0.0f;
    float c[10];
#pragma unroll
    for (int j = 0; j < 10; ++j) c[j] = 0.0f;
    __syncthreads();

    for (int u = -1; u < 200; ++u) {
        if (u >= 0) {
            const float* xpt = xprojdec + (size_t)u * 32 * 1200;
            for (int tt = wave; tt < 150; tt += 32) {
                int m0 = (tt & 1) * 16, n0 = (tt >> 1) * 16;
                v8f acc = {};
#pragma unroll
                for (int kt = 0; kt < 14; ++kt) {
                    v16bf a = frag_ld(hc, m0, 448, kt * 32, lane);
                    v16bf b = frag_ld(decW, n0, 448, kt * 32, lane);
                    acc = __builtin_amdgcn_wmma_f32_16x16x32_bf16(
                        false, a, false, b, (short)0, acc, false, false);
                }
                int n = n0 + (lane & 15);
                int mbase = m0 + ((lane & 16) ? 8 : 0);
#pragma unroll
                for (int r = 0; r < 8; ++r) {
                    int m = mbase + r;
                    gates[m * 1200 + n] = acc[r] + xpt[m * 1200 + n];
                }
            }
            __threadfence();
            __syncthreads();
            __bf16* lcu = lc + (size_t)u * 32 * 448;
#pragma unroll
            for (int j = 0; j < 10; ++j) {
                int i = tid + j * 1024;
                if (i < 9600) {
                    int b = i / 300, e = i - b * 300;
                    const float* g = gates + b * 1200;
                    float ig = sigm(g[e]);
                    float fg = sigm(g[300 + e]);
                    float gg = tanhf(g[600 + e]);
                    float og = sigm(g[900 + e]);
                    float cn = fg * c[j] + ig * gg;
                    c[j] = cn;
                    float hn = og * tanhf(cn);
                    hc[b * 448 + e] = (__bf16)hn;
                    lcu[b * 448 + e] = (__bf16)hn;
                }
            }
            // logits use the ctx that fed this step (still in ctx LDS)
            for (int i = tid; i < 4096; i += 1024) {
                int b = i >> 7, v = i & 127;
                lcu[b * 448 + 320 + v] = (__bf16)ctx[i];
            }
            __syncthreads();
        }
        // ---- attention: ctx = attend(h) ----
        if (wave < 16) {  // q[32,128] = h @ phiW^T + phib (wave-uniform branch)
            int m0 = (wave & 1) * 16, n0 = (wave >> 1) * 16;
            v8f acc = {};
#pragma unroll
            for (int kt = 0; kt < 10; ++kt) {
                v16bf a = frag_ld(hc, m0, 448, kt * 32, lane);
                v16bf b = frag_ld(phiW, n0, 320, kt * 32, lane);
                acc = __builtin_amdgcn_wmma_f32_16x16x32_bf16(
                    false, a, false, b, (short)0, acc, false, false);
            }
            int n = n0 + (lane & 15);
            float bias = phib[n];
            int mbase = m0 + ((lane & 16) ? 8 : 0);
#pragma unroll
            for (int r = 0; r < 8; ++r) qs[(mbase + r) * 128 + n] = acc[r] + bias;
        }
        __syncthreads();
        for (int i = tid; i < 4096; i += 1024) {  // scores[b,t] = q . keys
            int b = i >> 7, t = i & 127;
            const float* kp = keys + ((size_t)t * 32 + b) * 128;
            float s = 0.0f;
            for (int k = 0; k < 128; ++k) s += qs[b * 128 + k] * kp[k];
            sc[i] = s;
        }
        __syncthreads();
        if (tid < 32) {  // softmax over T_enc=128 per batch row
            float mx = -1e30f;
            for (int t = 0; t < 128; ++t) mx = fmaxf(mx, sc[tid * 128 + t]);
            float sum = 0.0f;
            for (int t = 0; t < 128; ++t) {
                float e = __expf(sc[tid * 128 + t] - mx);
                sc[tid * 128 + t] = e;
                sum += e;
            }
            float inv = 1.0f / sum;
            for (int t = 0; t < 128; ++t) sc[tid * 128 + t] *= inv;
        }
        __syncthreads();
        for (int i = tid; i < 4096; i += 1024) {  // ctx[b,v] = attn . values
            int b = i >> 7, v = i & 127;
            float s = 0.0f;
            for (int t = 0; t < 128; ++t)
                s += sc[b * 128 + t] * values[((size_t)t * 32 + b) * 128 + v];
            ctx[i] = s;
            hc[b * 448 + 320 + v] = (__bf16)s;
        }
        __syncthreads();
    }
}

// ---------------------------------------------------------------------------
extern "C" void kernel_launch(void* const* d_in, const int* in_sizes, int n_in,
                              void* d_out, int out_size, void* d_ws, size_t ws_size,
                              hipStream_t stream) {
    (void)in_sizes; (void)n_in; (void)out_size; (void)ws_size;
    const float* X = (const float*)d_in[0];
    const int*   Y = (const int*)d_in[2];
    const float* proj1_W = (const float*)d_in[43];
    const float* proj1_b = (const float*)d_in[44];
    const float* proj2_W = (const float*)d_in[45];
    const float* proj2_b = (const float*)d_in[46];
    const float* phi_W   = (const float*)d_in[47];
    const float* phi_b   = (const float*)d_in[48];
    const float* emb     = (const float*)d_in[49];
    const float* cellWih = (const float*)d_in[52];
    const float* cellWhh = (const float*)d_in[53];
    const float* cellbih = (const float*)d_in[54];
    const float* cellbhh = (const float*)d_in[55];
    const float* proj_W  = (const float*)d_in[56];
    const float* proj_b  = (const float*)d_in[57];

    char* wsb = (char*)d_ws;
    size_t off = 0;
    auto carve = [&](size_t bytes) {
        void* p = wsb + off;
        off += (bytes + 255) & ~(size_t)255;
        return p;
    };
    float*  xproj  = (float*)carve(2ull * 32768 * 1200 * 4);
    float*  gates  = (float*)carve(2ull * 32 * 1200 * 4);
    float*  keysB  = (float*)carve(4096ull * 128 * 4);
    float*  valsB  = (float*)carve(4096ull * 128 * 4);
    float*  xprojd = (float*)carve(6400ull * 1200 * 4);
    __bf16* actA   = (__bf16*)carve(1024ull * 32 * 608 * 2);
    __bf16* actB   = (__bf16*)carve(512ull * 32 * 1216 * 2);
    __bf16* xBf    = (__bf16*)carve(32768ull * 64 * 2);
    __bf16* embX   = (__bf16*)carve(6400ull * 416 * 2);
    __bf16* lcB    = (__bf16*)carve(6400ull * 448 * 2);
    __bf16* wihF   = (__bf16*)carve(1280ull * 1216 * 2);
    __bf16* wihB   = (__bf16*)carve(1280ull * 1216 * 2);
    __bf16* whhF   = (__bf16*)carve(1200ull * 320 * 2);
    __bf16* whhB   = (__bf16*)carve(1200ull * 320 * 2);
    __bf16* p1Bf   = (__bf16*)carve(128ull * 608 * 2);
    __bf16* p2Bf   = (__bf16*)carve(128ull * 608 * 2);
    __bf16* phiBf  = (__bf16*)carve(128ull * 320 * 2);
    __bf16* decW   = (__bf16*)carve(1200ull * 448 * 2);
    __bf16* wihE   = (__bf16*)carve(1280ull * 416 * 2);
    __bf16* pwBf   = (__bf16*)carve(128ull * 448 * 2);

    auto pk = [&](const float* s, int lds_, int co, int R, int K, __bf16* d,
                  int ldd, int doff, int Rp, int Kc) {
        int total = Rp * Kc;
        pack_bf16_kernel<<<dim3((total + 255) / 256), 256, 0, stream>>>(
            s, lds_, co, R, K, d, ldd, doff, Rp, Kc);
    };

    // one-shot packs (W rows padded to the GEMM grid's N coverage)
    pk(X, 40, 0, 32768, 40, xBf, 64, 0, 32768, 64);
    pk(proj1_W, 600, 0, 128, 600, p1Bf, 608, 0, 128, 608);
    pk(proj2_W, 600, 0, 128, 600, p2Bf, 608, 0, 128, 608);
    pk(phi_W, 300, 0, 128, 300, phiBf, 320, 0, 128, 320);
    pk(cellWhh, 300, 0, 1200, 300, decW, 448, 0, 1200, 320);      // [Whh|pad]
    pk(cellWih, 528, 400, 1200, 128, decW, 448, 320, 1200, 128);  // [Wih_ctx]
    pk(cellWih, 528, 0, 1200, 400, wihE, 416, 0, 1280, 416);      // emb part
    pk(proj_W, 428, 0, 33, 300, pwBf, 448, 0, 128, 320);          // h part
    pk(proj_W, 428, 300, 33, 128, pwBf, 448, 320, 128, 128);      // ctx part
    fill_zero_bf16_kernel<<<dim3(8192), 256, 0, stream>>>(actA, 1024ull * 32 * 608);
    fill_zero_bf16_kernel<<<dim3(4096), 256, 0, stream>>>(lcB, 6400ull * 448);

    // ---- Encoder: 4 pyramid BiLSTM layers ----
    const __bf16* cur = xBf;
    int T = 1024, D = 40, Kpad = 64;
    for (int L = 0; L < 4; ++L) {
        int base = 3 + L * 10;
        const float* Wih_f = (const float*)d_in[base + 0];
        const float* Whh_f = (const float*)d_in[base + 1];
        const float* bih_f = (const float*)d_in[base + 2];
        const float* bhh_f = (const float*)d_in[base + 3];
        const float* Wih_b = (const float*)d_in[base + 4];
        const float* Whh_b = (const float*)d_in[base + 5];
        const float* bih_b = (const float*)d_in[base + 6];
        const float* bhh_b = (const float*)d_in[base + 7];
        pk(Wih_f, D, 0, 1200, D, wihF, Kpad, 0, 1280, Kpad);
        pk(Wih_b, D, 0, 1200, D, wihB, Kpad, 0, 1280, Kpad);
        pk(Whh_f, 300, 0, 1200, 300, whhF, 320, 0, 1200, 320);
        pk(Whh_b, 300, 0, 1200, 300, whhB, 320, 0, 1200, 320);
        int M = T * 32;
        size_t dstride = (size_t)M * 1200;
        gemm_bias_kernel<<<dim3(10, M / 64), 256, 0, stream>>>(
            cur, Kpad, wihF, Kpad, bih_f, bhh_f, xproj, 1200, M, 1200, Kpad / 32);
        gemm_bias_kernel<<<dim3(10, M / 64), 256, 0, stream>>>(
            cur, Kpad, wihB, Kpad, bih_b, bhh_b, xproj + dstride, 1200, M, 1200, Kpad / 32);
        lstm_seq_kernel<<<2, 1024, 0, stream>>>(xproj, dstride, whhF, whhB,
                                                actA, gates, T);
        if (L < 3) {
            int Th = T / 2;
            pyramid_kernel<<<dim3((Th * 32 * 1216 + 255) / 256), 256, 0, stream>>>(
                actA, actB, Th);
            cur = actB; T = Th; D = 1200; Kpad = 1216;
        }
    }
    // keys / values projections: [4096,608] @ [128,608]^T
    gemm_bias_kernel<<<dim3(1, 64), 256, 0, stream>>>(
        actA, 608, p1Bf, 608, proj1_b, nullptr, keysB, 128, 4096, 128, 19);
    gemm_bias_kernel<<<dim3(1, 64), 256, 0, stream>>>(
        actA, 608, p2Bf, 608, proj2_b, nullptr, valsB, 128, 4096, 128, 19);

    // ---- Decoder ----
    embed_kernel<<<dim3(4096), 256, 0, stream>>>(Y, emb, embX);
    gemm_bias_kernel<<<dim3(10, 100), 256, 0, stream>>>(
        embX, 416, wihE, 416, cellbih, cellbhh, xprojd, 1200, 6400, 1200, 13);
    decoder_kernel<<<1, 1024, 0, stream>>>(xprojd, keysB, valsB, phiBf, phi_b,
                                           decW, lcB, gates);
    // logits: [6400,448] @ [128,448]^T + b -> d_out [6400,33] f32
    gemm_bias_kernel<<<dim3(1, 100), 256, 0, stream>>>(
        lcB, 448, pwBf, 448, proj_b, nullptr, (float*)d_out, 33, 6400, 33, 14);
}
